// CGCNN_58643483459784
// MI455X (gfx1250) — matrix operand order repaired and checked by reading, not compiled, not run on hardware
//
#include <hip/hip_runtime.h>
#include <hip/hip_bf16.h>
#include <math.h>

// ---------------- problem constants (match reference) ----------------
#define NN      50000        // nodes
#define NP      50048        // nodes padded to 64-row multiple (782*64)
#define EE      1600000      // edges (divisible by 64)
#define BB      500          // graphs
#define FIN     92           // orig_atom_fea_len
#define KP_EMB  96           // FIN padded to 32-multiple
#define DD      41           // nbr_fea_len
#define FF      64           // atom_fea_len
#define HH      128          // h_fea_len
#define NCONV   3
#define BN_EPS  1e-5f

// ---------------- WMMA plumbing (gfx1250, wave32) ----------------
typedef __bf16 bf16x16 __attribute__((ext_vector_type(16)));
typedef float  f32x8   __attribute__((ext_vector_type(8)));

union FragU { unsigned u[8]; bf16x16 v; };

__device__ __forceinline__ unsigned short f2bf(float f) {
    union { float f; unsigned u; } x; x.f = f;
    unsigned r = x.u + 0x7FFFu + ((x.u >> 16) & 1u);   // RNE truncate to bf16
    return (unsigned short)(r >> 16);
}

// A fragment: 16x32 bf16 tile, rows m0..m0+15, K slice k0..k0+31.
// LDS layout row-major [row][ldk] bf16. ISA layout: lane L (h=L>>4,m=L&15),
// vgpr v packs K = 16*(v>>2) + 8*h + 2*(v&3) (+1).
__device__ __forceinline__ bf16x16 load_fragA(const unsigned short* lds, int ldk,
                                              int m0, int k0, int lane) {
    int h = lane >> 4, m = lane & 15;
    const unsigned* p = (const unsigned*)(lds + (m0 + m) * ldk + k0);
    FragU f;
#pragma unroll
    for (int v = 0; v < 8; ++v) {
        int kb = 16 * (v >> 2) + 8 * h + 2 * (v & 3);
        f.u[v] = p[kb >> 1];
    }
    return f.v;
}

// B fragment: 32x16 bf16 tile (KxN), cols n0..n0+15, K slice k0..k0+31.
// LDS holds W transposed: [col][ldk] bf16.
// vgpr v packs K = 16*(v>>2) + 8*h + 2*(v&3) (+1), N = lane&15.
__device__ __forceinline__ bf16x16 load_fragB(const unsigned short* lds, int ldk,
                                              int n0, int k0, int lane) {
    int h = lane >> 4, n = lane & 15;
    const unsigned* p = (const unsigned*)(lds + (n0 + n) * ldk + k0);
    FragU f;
#pragma unroll
    for (int v = 0; v < 8; ++v) {
        int kb = 16 * (v >> 2) + 8 * h + 2 * (v & 3);
        f.u[v] = p[kb >> 1];
    }
    return f.v;
}

__device__ __forceinline__ f32x8 wmma_bf16(bf16x16 a, bf16x16 b, f32x8 c) {
    return __builtin_amdgcn_wmma_f32_16x16x32_bf16(false, a, false, b,
                                                   (short)0, c, false, false);
}

__device__ __forceinline__ float softplusf_(float x) {
    return (x > 20.f) ? x : log1pf(__expf(x));
}
__device__ __forceinline__ float sigmoidf_(float x) {
    return 1.f / (1.f + __expf(-x));
}

// ================= 1) embedding: h = x @ W_emb + b_emb =================
// Result staged through LDS -> fully coalesced b128 stores (h is padded to NP rows,
// so no store-side bounds checks; padded rows receive bias and are never consumed).
__global__ __launch_bounds__(128) void cg_embed(const float* __restrict__ x,
                                                const float* __restrict__ Wemb,
                                                const float* __restrict__ bemb,
                                                float* __restrict__ h) {
    __shared__ __align__(16) unsigned short As[64 * KP_EMB];   // 12 KB
    __shared__ __align__(16) unsigned short Bs[64 * KP_EMB];   // 12 KB
    __shared__ __align__(16) float Dt[64 * FF];                // 16 KB
    int tid = threadIdx.x;
    int row0 = blockIdx.x * 64;
    for (int idx = tid; idx < 64 * KP_EMB; idx += 128) {
        int r = idx / KP_EMB, k = idx % KP_EMB;
        int gr = row0 + r;
        float v = (gr < NN && k < FIN) ? x[gr * FIN + k] : 0.f;   // x is exactly NN rows
        As[idx] = f2bf(v);
    }
    for (int idx = tid; idx < 64 * KP_EMB; idx += 128) {
        int c = idx / KP_EMB, k = idx % KP_EMB;
        float v = (k < FIN) ? Wemb[k * FF + c] : 0.f;
        Bs[idx] = f2bf(v);
    }
    __syncthreads();
    int wave = tid >> 5, lane = tid & 31;
    int m0 = wave * 16;
#pragma unroll
    for (int ct = 0; ct < 4; ++ct) {
        f32x8 acc = {};
#pragma unroll
        for (int ks = 0; ks < 3; ++ks) {
            bf16x16 a = load_fragA(As, KP_EMB, m0, ks * 32, lane);
            bf16x16 b = load_fragB(Bs, KP_EMB, ct * 16, ks * 32, lane);
            acc = wmma_bf16(a, b, acc);
        }
        int hh = lane >> 4, n = lane & 15;
        int col = ct * 16 + n;
        float bias = bemb[col];
#pragma unroll
        for (int r = 0; r < 8; ++r)
            Dt[(m0 + r + 8 * hh) * FF + col] = acc[r] + bias;
    }
    __syncthreads();
    // coalesced float4 stores (unconditional: h has NP padded rows)
    const float4* D4 = (const float4*)Dt;
    float4* H4 = (float4*)(h + (size_t)row0 * FF);
    for (int idx = tid; idx < 64 * (FF / 4); idx += 128)
        H4[idx] = D4[idx];
}

// ====== 2) node partials: Pn[NP][256] = h @ [Wf_i|Wf_j|Ws_i|Ws_j] (+bias on *_i) ======
__global__ __launch_bounds__(128) void cg_nodepart(const float* __restrict__ h,
                                                   const float* __restrict__ Wf_l,
                                                   const float* __restrict__ Ws_l,
                                                   const float* __restrict__ bf_l,
                                                   const float* __restrict__ bs_l,
                                                   float* __restrict__ Pn) {
    __shared__ __align__(16) unsigned short As[64 * 64];    // 8 KB
    __shared__ __align__(16) unsigned short Bs[256 * 64];   // 32 KB
    int tid = threadIdx.x;
    int row0 = blockIdx.x * 64;
    for (int idx = tid; idx < 64 * 64; idx += 128) {
        int r = idx >> 6, k = idx & 63;
        As[idx] = f2bf(h[(row0 + r) * FF + k]);   // h padded: no guard
    }
    for (int idx = tid; idx < 256 * 64; idx += 128) {
        int c = idx >> 6, k = idx & 63;
        float v;
        if (c < 64)       v = Wf_l[k * FF + c];                // x_i rows of Wf
        else if (c < 128) v = Wf_l[(64 + k) * FF + (c - 64)];  // x_j rows of Wf
        else if (c < 192) v = Ws_l[k * FF + (c - 128)];        // x_i rows of Ws
        else              v = Ws_l[(64 + k) * FF + (c - 192)]; // x_j rows of Ws
        Bs[idx] = f2bf(v);
    }
    __syncthreads();
    int wave = tid >> 5, lane = tid & 31;
    int m0 = wave * 16;
    for (int ct = 0; ct < 16; ++ct) {
        f32x8 acc = {};
#pragma unroll
        for (int ks = 0; ks < 2; ++ks) {
            bf16x16 a = load_fragA(As, 64, m0, ks * 32, lane);
            bf16x16 b = load_fragB(Bs, 64, ct * 16, ks * 32, lane);
            acc = wmma_bf16(a, b, acc);
        }
        int hh = lane >> 4, n = lane & 15;
        int col = ct * 16 + n;
        float bias = (col < 64) ? bf_l[col]
                   : (col >= 128 && col < 192) ? bs_l[col - 128] : 0.f;
        int rbase = row0 + m0 + 8 * hh;
#pragma unroll
        for (int r = 0; r < 8; ++r)                 // unconditional: Pn padded
            Pn[(rbase + r) * 256 + col] = acc[r] + bias;
    }
}

// ====== 3) edge GEMM + gate + scatter-add (fused) ======
// D[64 edges][128] = e_attr @ [Wf_e | Ws_e], then
// m = sigmoid(D_f + Pf_i[dst] + Pf_j[src]) * softplus(D_s + Ps_i[dst] + Ps_j[src])
// atomically accumulated into agg[dst].
__global__ __launch_bounds__(128) void cg_edgemsg(const float* __restrict__ ea,
                                                  const float* __restrict__ Wf_l,
                                                  const float* __restrict__ Ws_l,
                                                  const float* __restrict__ Pn,
                                                  const int* __restrict__ src,
                                                  const int* __restrict__ dst,
                                                  float* __restrict__ agg) {
    __shared__ __align__(16) unsigned short As[64 * 64];    // 8 KB
    __shared__ __align__(16) unsigned short Bs[128 * 64];   // 16 KB
    __shared__ __align__(16) float Dt[64 * 128];            // 32 KB
    int tid = threadIdx.x;
    int e0 = blockIdx.x * 64;
    for (int idx = tid; idx < 64 * 64; idx += 128) {
        int r = idx >> 6, k = idx & 63;
        int e = e0 + r;
        As[idx] = (k < DD) ? f2bf(ea[e * DD + k]) : (unsigned short)0;
    }
    for (int idx = tid; idx < 128 * 64; idx += 128) {
        int c = idx >> 6, k = idx & 63;
        float v = 0.f;
        if (k < DD)
            v = (c < 64) ? Wf_l[(128 + k) * FF + c] : Ws_l[(128 + k) * FF + (c - 64)];
        Bs[idx] = f2bf(v);
    }
    __syncthreads();
    int wave = tid >> 5, lane = tid & 31;
    int m0 = wave * 16;
    for (int ct = 0; ct < 8; ++ct) {
        f32x8 acc = {};
#pragma unroll
        for (int ks = 0; ks < 2; ++ks) {
            bf16x16 a = load_fragA(As, 64, m0, ks * 32, lane);
            bf16x16 b = load_fragB(Bs, 64, ct * 16, ks * 32, lane);
            acc = wmma_bf16(a, b, acc);
        }
        int hh = lane >> 4, n = lane & 15;
#pragma unroll
        for (int r = 0; r < 8; ++r)
            Dt[(m0 + r + 8 * hh) * 128 + ct * 16 + n] = acc[r];
    }
    __syncthreads();
    // epilogue: coalesced gather of node partials, gate, scatter-add
    for (int idx = tid; idx < 64 * 64; idx += 128) {
        int r = idx >> 6, c = idx & 63;
        int e = e0 + r;
        int d = dst[e], s = src[e];
        float fpre = Dt[r * 128 + c]
                   + Pn[d * 256 + c] + Pn[s * 256 + 64 + c];
        float spre = Dt[r * 128 + 64 + c]
                   + Pn[d * 256 + 128 + c] + Pn[s * 256 + 192 + c];
        float m = sigmoidf_(fpre) * softplusf_(spre);
        atomicAdd(&agg[d * FF + c], m);
    }
}

// ====== 4) batch-norm statistics (sum / sumsq per feature) ======
__global__ __launch_bounds__(256) void cg_bnstat(const float* __restrict__ agg,
                                                 float* __restrict__ acc) {
    __shared__ float s[256], q[256];
    int tid = threadIdx.x;
    int c = tid & 63, rsub = tid >> 6;
    float su = 0.f, sq = 0.f;
    for (int r = blockIdx.x * 4 + rsub; r < NN; r += gridDim.x * 4) {
        float v = agg[r * FF + c];
        su += v; sq += v * v;
    }
    s[tid] = su; q[tid] = sq;
    __syncthreads();
    if (tid < 128) { s[tid] += s[tid + 128]; q[tid] += q[tid + 128]; }
    __syncthreads();
    if (tid < 64) {
        atomicAdd(&acc[c],      s[tid] + s[tid + 64]);
        atomicAdd(&acc[64 + c], q[tid] + q[tid + 64]);
    }
}

// ====== 5) BN apply + residual: h += gamma*(agg-mu)*rsqrt(var+eps) + beta ======
__global__ __launch_bounds__(256) void cg_bnapply(float* __restrict__ h,
                                                  const float* __restrict__ agg,
                                                  const float* __restrict__ acc,
                                                  const float* __restrict__ gamma,
                                                  const float* __restrict__ beta) {
    int i = blockIdx.x * blockDim.x + threadIdx.x;
    if (i >= NN * FF) return;
    int c = i & 63;
    const float invN = 1.f / (float)NN;
    float mu = acc[c] * invN;
    float var = acc[64 + c] * invN - mu * mu;
    float inv = rsqrtf(var + BN_EPS);
    h[i] += (agg[i] - mu) * inv * gamma[c] + beta[c];
}

// ====== 6) per-graph mean pool (sums + counts) ======
__global__ __launch_bounds__(256) void cg_pool(const float* __restrict__ h,
                                               const int* __restrict__ batch,
                                               float* __restrict__ pooled,
                                               float* __restrict__ cnt) {
    int i = blockIdx.x * blockDim.x + threadIdx.x;
    if (i >= NN * FF) return;
    int r = i >> 6, c = i & 63;
    int b = batch[r];
    atomicAdd(&pooled[b * FF + c], h[i]);
    if (c == 0) atomicAdd(&cnt[b], 1.f);
}

// ====== 7) head: softplus -> Linear(64,128) -> softplus -> Linear(128,1) ======
__global__ __launch_bounds__(128) void cg_head(const float* __restrict__ pooled,
                                               const float* __restrict__ cnt,
                                               const float* __restrict__ Wfc,
                                               const float* __restrict__ bfc,
                                               const float* __restrict__ Wout,
                                               const float* __restrict__ bout,
                                               float* __restrict__ out) {
    __shared__ float sp[64];
    __shared__ float red[128];
    int b = blockIdx.x, t = threadIdx.x;
    if (t < 64) {
        float cc = fmaxf(cnt[b], 1.f);
        sp[t] = softplusf_(pooled[b * FF + t] / cc);
    }
    __syncthreads();
    float a = bfc[t];
#pragma unroll 8
    for (int k = 0; k < 64; ++k) a += sp[k] * Wfc[k * HH + t];
    red[t] = softplusf_(a) * Wout[t];
    __syncthreads();
    for (int s = 64; s > 0; s >>= 1) {
        if (t < s) red[t] += red[t + s];
        __syncthreads();
    }
    if (t == 0) out[b] = red[0] + bout[0];
}

// ================================ launcher ================================
extern "C" void kernel_launch(void* const* d_in, const int* in_sizes, int n_in,
                              void* d_out, int out_size, void* d_ws, size_t ws_size,
                              hipStream_t stream) {
    (void)in_sizes; (void)n_in; (void)out_size; (void)ws_size;
    const float* x     = (const float*)d_in[0];
    const int*   eidx  = (const int*)  d_in[1];
    const float* ea    = (const float*)d_in[2];
    const int*   batch = (const int*)  d_in[3];
    const float* Wemb  = (const float*)d_in[4];
    const float* bemb  = (const float*)d_in[5];
    const float* Wf    = (const float*)d_in[6];
    const float* bf    = (const float*)d_in[7];
    const float* Ws    = (const float*)d_in[8];
    const float* bs    = (const float*)d_in[9];
    const float* gamma = (const float*)d_in[10];
    const float* beta  = (const float*)d_in[11];
    const float* Wfc   = (const float*)d_in[12];
    const float* bfc   = (const float*)d_in[13];
    const float* Wout  = (const float*)d_in[14];
    const float* bout  = (const float*)d_in[15];

    const int* src = eidx;        // edge_index[0]  (x_j = h[src])
    const int* dst = eidx + EE;   // edge_index[1]  (x_i = h[dst], scatter target)

    float* ws     = (float*)d_ws;
    float* h      = ws;                                   // NP*64 (padded)
    float* Pn     = h + (size_t)NP * FF;                  // NP*256 (padded)
    float* agg    = Pn + (size_t)NP * 256;                // NP*64 (padded)
    float* bnAcc  = agg + (size_t)NP * FF;                // 128
    float* pooled = bnAcc + 128;                          // B*64
    float* cnt    = pooled + (size_t)BB * FF;             // B

    const int nblk = NP / 64;   // 782

    cg_embed<<<nblk, 128, 0, stream>>>(x, Wemb, bemb, h);

    for (int l = 0; l < NCONV; ++l) {
        const float* Wf_l = Wf + (size_t)l * (2 * FF + DD) * FF;
        const float* Ws_l = Ws + (size_t)l * (2 * FF + DD) * FF;
        cg_nodepart<<<nblk, 128, 0, stream>>>(h, Wf_l, Ws_l,
                                              bf + l * FF, bs + l * FF, Pn);
        hipMemsetAsync(agg, 0, (size_t)NP * FF * sizeof(float), stream);
        hipMemsetAsync(bnAcc, 0, 128 * sizeof(float), stream);
        cg_edgemsg<<<EE / 64, 128, 0, stream>>>(ea, Wf_l, Ws_l, Pn, src, dst, agg);
        cg_bnstat<<<256, 256, 0, stream>>>(agg, bnAcc);
        cg_bnapply<<<(NN * FF + 255) / 256, 256, 0, stream>>>(h, agg, bnAcc,
                                                              gamma + l * FF,
                                                              beta + l * FF);
    }

    hipMemsetAsync(pooled, 0, ((size_t)BB * FF + BB) * sizeof(float), stream);
    cg_pool<<<(NN * FF + 255) / 256, 256, 0, stream>>>(h, batch, pooled, cnt);
    cg_head<<<BB, 128, 0, stream>>>(pooled, cnt, Wfc, bfc, Wout, bout, (float*)d_out);
}
